// PrefSimMat_82145544504102
// MI455X (gfx1250) — matrix operand order repaired and checked by reading, not compiled, not run on hardware
//
#include <hip/hip_runtime.h>
#include <hip/hip_bf16.h>

// ---------------------------------------------------------------------------
// PrefSimMat 'EucDis' on MI455X (gfx1250, wave32, WMMA):
//   sim = 1 - dist / row_l2norm(dist),  dist_ij = sqrt(max(s_i + s_j - 2*G, 0))
// Row norms computed analytically (N*s_i + S - 2*x_i.T) -> single fused pass
// over the 256 MB output (write floor ~11us @ 23.3 TB/s).
// Gram via one-shot bf16 hi/lo split (bf16x3, near-f32 accuracy) feeding
// v_wmma_f32_16x16x32_bf16; K staged in double-buffered LDS chunks via
// async global->LDS copies (ASYNCcnt) overlapped with WMMA compute.
// ---------------------------------------------------------------------------

#define NROWS 8192
#define DCOLS 256
#define BM 128
#define BN 128
#define BK 32      // K chunk; 4 arrays x 2 buffers x 128x32 bf16 = 64 KB LDS

typedef __bf16 v16bf __attribute__((ext_vector_type(16)));
typedef float  v8f   __attribute__((ext_vector_type(8)));
typedef int    v4i   __attribute__((vector_size(16)));   // matches builtin proto

union Frag {
    uint4 u[2];
    v16bf v;
};

#if defined(__AMDGCN__) && \
    __has_builtin(__builtin_amdgcn_global_load_async_to_lds_b128) && \
    __has_builtin(__builtin_amdgcn_s_wait_asynccnt)
#define USE_ASYNC 1
#else
#define USE_ASYNC 0
#endif

#if USE_ASYNC
__device__ __forceinline__ void async_copy_b128(const void* src, void* dstLds) {
    // generic->AS1 / generic->AS3 via integer cast; low 32 bits of a generic
    // LDS pointer are the LDS offset per the CDNA5 aperture mapping.
    // Builtin proto (from clang-22 diagnostic): (v4i __device__*, v4i __shared__*, Ii, Ii)
    __builtin_amdgcn_global_load_async_to_lds_b128(
        (__attribute__((address_space(1))) v4i*)(unsigned long long)src,
        (__attribute__((address_space(3))) v4i*)(unsigned)(unsigned long long)dstLds,
        0, 0);
}
#endif

__device__ __forceinline__ unsigned short f2bf(float f) {
    unsigned u = __float_as_uint(f);
    u += 0x7FFFu + ((u >> 16) & 1u);   // round-to-nearest-even (finite inputs)
    return (unsigned short)(u >> 16);
}
__device__ __forceinline__ float bf2f(unsigned short h) {
    return __uint_as_float(((unsigned)h) << 16);
}

__device__ __forceinline__ void split8(const float4 a, const float4 b,
                                       uint4& hu, uint4& lu) {
    float v[8] = {a.x, a.y, a.z, a.w, b.x, b.y, b.z, b.w};
    unsigned short h[8], l[8];
#pragma unroll
    for (int i = 0; i < 8; ++i) {
        h[i] = f2bf(v[i]);
        l[i] = f2bf(v[i] - bf2f(h[i]));
    }
    hu.x = (unsigned)h[0] | ((unsigned)h[1] << 16);
    hu.y = (unsigned)h[2] | ((unsigned)h[3] << 16);
    hu.z = (unsigned)h[4] | ((unsigned)h[5] << 16);
    hu.w = (unsigned)h[6] | ((unsigned)h[7] << 16);
    lu.x = (unsigned)l[0] | ((unsigned)l[1] << 16);
    lu.y = (unsigned)l[2] | ((unsigned)l[3] << 16);
    lu.z = (unsigned)l[4] | ((unsigned)l[5] << 16);
    lu.w = (unsigned)l[6] | ((unsigned)l[7] << 16);
}

// ---------------------------------------------------------------------------
// Pass 0: one-shot f32 -> bf16 hi/lo split (removes 64x redundant per-block
// conversion; the 8 MB of panels stay resident in the 192 MB L2).
// ---------------------------------------------------------------------------
__global__ __launch_bounds__(256) void
prefsim_split(const float* __restrict__ x, unsigned short* __restrict__ xhi,
              unsigned short* __restrict__ xlo) {
    const size_t g = (size_t)blockIdx.x * 256 + threadIdx.x;  // 8-elem group
    const float* p = x + g * 8;
    float4 a = *(const float4*)p;
    float4 b = *(const float4*)(p + 4);
    uint4 hu, lu;
    split8(a, b, hu, lu);
    *(uint4*)(xhi + g * 8) = hu;
    *(uint4*)(xlo + g * 8) = lu;
}

// ---------------------------------------------------------------------------
// Pass 1: s_i = ||x_i||^2 ; T = sum_i x_i ; S = sum_i s_i
// ---------------------------------------------------------------------------
__global__ __launch_bounds__(256) void
prefsim_stats(const float* __restrict__ x, float* __restrict__ s,
              float* __restrict__ T, float* __restrict__ Ssum) {
    __shared__ float tLds[DCOLS];
    const int t = threadIdx.x;
    tLds[t] = 0.f;
    __syncthreads();

    const int wave = t >> 5, lane = t & 31;
    const int rowBase = blockIdx.x * 32 + wave * 4;

    float colAcc[8];
#pragma unroll
    for (int i = 0; i < 8; ++i) colAcc[i] = 0.f;
    float sBlk = 0.f;

    for (int rr = 0; rr < 4; ++rr) {
        const int row = rowBase + rr;
        const float* p = x + (size_t)row * DCOLS + lane * 8;
        float4 a = *(const float4*)p;
        float4 b = *(const float4*)(p + 4);
        colAcc[0] += a.x; colAcc[1] += a.y; colAcc[2] += a.z; colAcc[3] += a.w;
        colAcc[4] += b.x; colAcc[5] += b.y; colAcc[6] += b.z; colAcc[7] += b.w;
        float sq = a.x * a.x + a.y * a.y + a.z * a.z + a.w * a.w
                 + b.x * b.x + b.y * b.y + b.z * b.z + b.w * b.w;
#pragma unroll
        for (int off = 16; off > 0; off >>= 1) sq += __shfl_down(sq, off, 32);
        if (lane == 0) { s[row] = sq; sBlk += sq; }
    }
#pragma unroll
    for (int i = 0; i < 8; ++i) atomicAdd(&tLds[lane * 8 + i], colAcc[i]);  // ds_add_f32
    __syncthreads();
    atomicAdd(&T[t], tLds[t]);
    if (lane == 0) atomicAdd(Ssum, sBlk);
}

// ---------------------------------------------------------------------------
// Pass 2: inv_rn_i = 1 / max(sqrt(N*s_i + S - 2*x_i.T), eps). One wave/row.
// ---------------------------------------------------------------------------
__global__ __launch_bounds__(256) void
prefsim_rownorm(const float* __restrict__ x, const float* __restrict__ s,
                const float* __restrict__ T, const float* __restrict__ Ssum,
                float* __restrict__ inv_rn) {
    const int t = threadIdx.x;
    const int wave = t >> 5, lane = t & 31;
    const int row = blockIdx.x * 8 + wave;

    const float* p = x + (size_t)row * DCOLS + lane * 8;
    const float* q = T + lane * 8;
    float4 a  = *(const float4*)p, b  = *(const float4*)(p + 4);
    float4 ta = *(const float4*)q, tb = *(const float4*)(q + 4);
    float dot = a.x * ta.x + a.y * ta.y + a.z * ta.z + a.w * ta.w
              + b.x * tb.x + b.y * tb.y + b.z * tb.z + b.w * tb.w;
#pragma unroll
    for (int off = 16; off > 0; off >>= 1) dot += __shfl_down(dot, off, 32);
    if (lane == 0) {
        float v  = fmaxf((float)NROWS * s[row] + *Ssum - 2.f * dot, 0.f);
        float rn = sqrtf(v);
        inv_rn[row] = 1.f / fmaxf(rn, 1e-12f);
    }
}

// ---------------------------------------------------------------------------
// Stage one K chunk of the hi/lo A and B panels into (swizzled) LDS.
// Swizzle key (row>>2)&3: rows aliasing at the 64 B LDS row stride land in
// distinct 16 B chunks -> conflict-free ds_load_b128 fragment reads.
// ---------------------------------------------------------------------------
__device__ __forceinline__ void
stage_chunk(const unsigned short* __restrict__ xhi,
            const unsigned short* __restrict__ xlo,
            int rowBlk, int colBlk, int ck, int t,
            unsigned short* ahi, unsigned short* alo,
            unsigned short* bhi, unsigned short* blo) {
#pragma unroll
    for (int i = 0; i < 2; ++i) {
        const int g   = i * 256 + t;   // 512 groups of 8 bf16 per array
        const int row = g >> 2;        // 0..127 panel-local
        const int c8  = g & 3;         // 8-elem chunk within BK=32
        const size_t srcA = (size_t)(rowBlk + row) * DCOLS + ck * BK + (c8 << 3);
        const size_t srcB = (size_t)(colBlk + row) * DCOLS + ck * BK + (c8 << 3);
        const int dst = row * BK + ((c8 ^ ((row >> 2) & 3)) << 3);
#if USE_ASYNC
        async_copy_b128(xhi + srcA, ahi + dst);
        async_copy_b128(xlo + srcA, alo + dst);
        async_copy_b128(xhi + srcB, bhi + dst);
        async_copy_b128(xlo + srcB, blo + dst);
#else
        *(uint4*)(ahi + dst) = *(const uint4*)(xhi + srcA);
        *(uint4*)(alo + dst) = *(const uint4*)(xlo + srcA);
        *(uint4*)(bhi + dst) = *(const uint4*)(xhi + srcB);
        *(uint4*)(blo + dst) = *(const uint4*)(xlo + srcB);
#endif
    }
}

// ---------------------------------------------------------------------------
// Pass 3: fused Gram + distance + normalize. 128x128 block / 8 waves, each
// wave 64x32 (eight 16x16 f32 accumulators), 24 WMMAs per 32-K chunk,
// double-buffered async LDS pipeline.
// ---------------------------------------------------------------------------
__global__ __launch_bounds__(256) void
prefsim_gemm(const unsigned short* __restrict__ xhi,
             const unsigned short* __restrict__ xlo,
             const float* __restrict__ s, const float* __restrict__ inv_rn,
             float* __restrict__ out) {
    __shared__ __align__(16) unsigned short AhiS[2][BM * BK];
    __shared__ __align__(16) unsigned short AloS[2][BM * BK];
    __shared__ __align__(16) unsigned short BhiS[2][BN * BK];
    __shared__ __align__(16) unsigned short BloS[2][BN * BK];

    const int t     = threadIdx.x;
    const int wid   = t >> 5;
    const int lane  = t & 31;
    const int half  = lane >> 4;
    const int l15   = lane & 15;
    const int waveM = wid & 1;    // 2 waves over M (64 rows each)
    const int waveN = wid >> 1;   // 4 waves over N (32 cols each)
    const int rowBlk = blockIdx.y * BM;
    const int colBlk = blockIdx.x * BN;

    v8f acc[4][2];
#pragma unroll
    for (int mt = 0; mt < 4; ++mt)
#pragma unroll
        for (int nt = 0; nt < 2; ++nt)
#pragma unroll
            for (int r = 0; r < 8; ++r) acc[mt][nt][r] = 0.f;

#if USE_ASYNC
    stage_chunk(xhi, xlo, rowBlk, colBlk, 0, t,
                AhiS[0], AloS[0], BhiS[0], BloS[0]);
    __builtin_amdgcn_s_wait_asynccnt(0);
    __syncthreads();
#endif

    for (int ck = 0; ck < DCOLS / BK; ++ck) {
        const int cur = ck & 1;
#if USE_ASYNC
        if (ck < DCOLS / BK - 1)   // copy next chunk while computing this one
            stage_chunk(xhi, xlo, rowBlk, colBlk, ck + 1, t,
                        AhiS[cur ^ 1], AloS[cur ^ 1], BhiS[cur ^ 1], BloS[cur ^ 1]);
#else
        __syncthreads();
        stage_chunk(xhi, xlo, rowBlk, colBlk, ck, t,
                    AhiS[cur], AloS[cur], BhiS[cur], BloS[cur]);
        __syncthreads();
#endif
        // ---- B fragments: 16 contiguous K per half-wave (K = half*16 + j)
        Frag bh[2], bl[2];
#pragma unroll
        for (int nt = 0; nt < 2; ++nt) {
            const int col = waveN * 32 + nt * 16 + l15;
            const int sw  = (col >> 2) & 3;
            const int c0  = half * 2;
            const unsigned short* bp = BhiS[cur] + col * BK;
            const unsigned short* bq = BloS[cur] + col * BK;
            bh[nt].u[0] = *(const uint4*)(bp + (((c0    ) ^ sw) << 3));
            bh[nt].u[1] = *(const uint4*)(bp + (((c0 + 1) ^ sw) << 3));
            bl[nt].u[0] = *(const uint4*)(bq + (((c0    ) ^ sw) << 3));
            bl[nt].u[1] = *(const uint4*)(bq + (((c0 + 1) ^ sw) << 3));
        }
#pragma unroll
        for (int mt = 0; mt < 4; ++mt) {
            // A fragments: 16 contiguous K per half-wave (K = half*16 + j)
            const int rowL = waveM * 64 + mt * 16 + l15;
            const int sw   = (rowL >> 2) & 3;
            const int c0   = half * 2;
            const int c1   = half * 2 + 1;
            Frag ah, al;
            const unsigned short* ap = AhiS[cur] + rowL * BK;
            const unsigned short* aq = AloS[cur] + rowL * BK;
            ah.u[0] = *(const uint4*)(ap + ((c0 ^ sw) << 3));
            ah.u[1] = *(const uint4*)(ap + ((c1 ^ sw) << 3));
            al.u[0] = *(const uint4*)(aq + ((c0 ^ sw) << 3));
            al.u[1] = *(const uint4*)(aq + ((c1 ^ sw) << 3));
#pragma unroll
            for (int nt = 0; nt < 2; ++nt) {
                acc[mt][nt] = __builtin_amdgcn_wmma_f32_16x16x32_bf16(
                    false, ah.v, false, bh[nt].v, (short)0, acc[mt][nt], false, false);
                acc[mt][nt] = __builtin_amdgcn_wmma_f32_16x16x32_bf16(
                    false, ah.v, false, bl[nt].v, (short)0, acc[mt][nt], false, false);
                acc[mt][nt] = __builtin_amdgcn_wmma_f32_16x16x32_bf16(
                    false, al.v, false, bh[nt].v, (short)0, acc[mt][nt], false, false);
            }
        }
#if USE_ASYNC
        __builtin_amdgcn_s_wait_asynccnt(0);
        __syncthreads();
#endif
    }

    // ---- fused epilogue: sq -> dist -> 1 - dist*inv_rn, NT streaming stores
#pragma unroll
    for (int mt = 0; mt < 4; ++mt) {
        const int rBase = rowBlk + waveM * 64 + mt * 16 + half * 8;  // C: M = r + 8*half
        float sr[8], ir[8];
#pragma unroll
        for (int r = 0; r < 8; ++r) { sr[r] = s[rBase + r]; ir[r] = inv_rn[rBase + r]; }
#pragma unroll
        for (int nt = 0; nt < 2; ++nt) {
            const int col = colBlk + waveN * 32 + nt * 16 + l15;     // C: N = lane&15
            const float sc = s[col];
#pragma unroll
            for (int r = 0; r < 8; ++r) {
                float sq = sr[r] + sc - 2.f * acc[mt][nt][r];
                float d  = (sq > 0.f) ? sqrtf(sq) : 0.f;
                float v  = 1.f - d * ir[r];
                __builtin_nontemporal_store(v, out + (size_t)(rBase + r) * NROWS + col);
            }
        }
    }
}

// ---------------------------------------------------------------------------
extern "C" void kernel_launch(void* const* d_in, const int* in_sizes, int n_in,
                              void* d_out, int out_size, void* d_ws, size_t ws_size,
                              hipStream_t stream) {
    const float* x = (const float*)d_in[0];
    float* out = (float*)d_out;

    float* s      = (float*)d_ws;            // 8192 f32
    float* inv_rn = s + NROWS;               // 8192 f32
    float* T      = s + 2 * NROWS;           // 256 f32
    float* Ssum   = T + DCOLS;               // 1 f32
    unsigned short* Xhi = (unsigned short*)(s + 16896);          // 4 MB bf16 hi
    unsigned short* Xlo = Xhi + (size_t)NROWS * DCOLS;           // 4 MB bf16 lo

    (void)hipMemsetAsync(T, 0, (DCOLS + 1) * sizeof(float), stream);  // capture-safe

    prefsim_split<<<NROWS * DCOLS / (8 * 256), 256, 0, stream>>>(x, Xhi, Xlo);
    prefsim_stats<<<NROWS / 32, 256, 0, stream>>>(x, s, T, Ssum);
    prefsim_rownorm<<<NROWS / 8, 256, 0, stream>>>(x, s, T, Ssum, inv_rn);

    dim3 grid(NROWS / BN, NROWS / BM);
    prefsim_gemm<<<grid, 256, 0, stream>>>(Xhi, Xlo, s, inv_rn, out);
}
